// Detector_head_84430467104978
// MI455X (gfx1250) — matrix operand order repaired and checked by reading, not compiled, not run on hardware
//
#include <hip/hip_runtime.h>

// ---------------------------------------------------------------- constants
#define NB      32
#define CIN     128
#define CHID    256
#define COUT    65
#define IH      60
#define IW      80
#define NP      4800        // 60*80
#define PH      480
#define PW      640
#define NPIX    9830400     // 32*480*640
#define TOPK    1000u

#define LOGITS_N 9984000    // 32*65*4800

// padded bf16 input: [32][64 ch-pairs][62][88][2]
#define XPADY   62
#define XPADX   88
#define XPLANE  5456        // 62*88 (u32 units per ch-pair plane)

typedef __attribute__((ext_vector_type(16))) __bf16 v16bf;
typedef __attribute__((ext_vector_type(8)))  __bf16 v8bf;
typedef __attribute__((ext_vector_type(8)))  float  v8f;

// workspace byte offsets
#define WPK1_OFF   0u           // 9*4*256*32 bf16 = 589824 B
#define W2PK_OFF   589824u      // 128*256 bf16    = 65536 B
#define S1_OFF     720896u      // 256 f32
#define B1_OFF     721920u      // 256 f32
#define S2_OFF     722944u      // 96 f32
#define B2_OFF     723328u      // 96 f32
#define SEL_OFF    724992u      // selHi | kRem | selLo | kRem2 | kth (128B each)
#define H_OFF      786432u      // h bf16 [32][4800][256] = 78643200 B -> ends 79429632
#define HMAX_OFF   H_OFF        // reuse: h dead after conv2
#define HIST_OFF   (H_OFF + 40000000u)  // 8388608 B, inside h span
#define XPAD_OFF   80000000u    // 32*64*5456*4 = 44695552 B

// ---------------------------------------------------------------- async global->LDS (guarded)
#if defined(__AMDGCN__) && __has_builtin(__builtin_amdgcn_global_load_async_to_lds_b128) && \
    __has_builtin(__builtin_amdgcn_global_load_async_to_lds_b32) && \
    __has_builtin(__builtin_amdgcn_s_wait_asynccnt)
#define USE_ASYNC 1
#else
#define USE_ASYNC 0
#endif

#if USE_ASYNC
typedef int i32v4 __attribute__((vector_size(16)));
__device__ __forceinline__ void g2l128(const void* g, void* l) {
    __builtin_amdgcn_global_load_async_to_lds_b128(
        (__attribute__((address_space(1))) i32v4*)g,
        (__attribute__((address_space(3))) i32v4*)l, 0, 0);
}
__device__ __forceinline__ void g2l32(const void* g, void* l) {
    __builtin_amdgcn_global_load_async_to_lds_b32(
        (__attribute__((address_space(1))) int*)g,
        (__attribute__((address_space(3))) int*)l, 0, 0);
}
#endif

// ---------------------------------------------------------------- helpers
__device__ __forceinline__ unsigned short f2bf(float f) {
    unsigned u = __float_as_uint(f);
    unsigned r = u + 0x7FFFu + ((u >> 16) & 1u);   // round-to-nearest-even
    return (unsigned short)(r >> 16);
}
__device__ __forceinline__ unsigned orderable(float f) {
    unsigned u = __float_as_uint(f);
    return (u >> 31) ? ~u : (u | 0x80000000u);
}
__device__ __forceinline__ float fromOrderable(unsigned u) {
    return (u >> 31) ? __uint_as_float(u & 0x7FFFFFFFu) : __uint_as_float(~u);
}

// ---------------------------------------------------------------- prep: pack weights, fold BN
__global__ __launch_bounds__(256) void prep_k(
    const float* __restrict__ wa, const float* __restrict__ ba,
    const float* __restrict__ ga, const float* __restrict__ bta,
    const float* __restrict__ ma, const float* __restrict__ va,
    const float* __restrict__ wb, const float* __restrict__ bbv,
    const float* __restrict__ gb, const float* __restrict__ btb,
    const float* __restrict__ mbm, const float* __restrict__ vbv,
    unsigned short* __restrict__ wpk1, unsigned short* __restrict__ w2pk,
    float* __restrict__ s1, float* __restrict__ b1,
    float* __restrict__ s2, float* __restrict__ b2)
{
    int idx = blockIdx.x * 256 + threadIdx.x;
    if (idx < 294912) {                 // wa: [256][128][9] flat == idx
        int oc = idx / 1152, rem = idx % 1152;
        int ic = rem / 9,    tap = rem % 9;
        wpk1[(((tap * 4 + (ic >> 5)) * 256 + oc) * 32) + (ic & 31)] = f2bf(wa[idx]);
    } else if (idx < 294912 + 32768) {  // wb -> [128][256], rows >=65 zero
        int j = idx - 294912;
        int m = j >> 8;
        float v = (m < 65) ? wb[j] : 0.0f;
        w2pk[j] = f2bf(v);
    } else if (idx < 294912 + 32768 + 256) {
        int c = idx - (294912 + 32768);
        float s = ga[c] * rsqrtf(va[c] + 1e-5f);
        s1[c] = s;
        b1[c] = ba[c] * s + bta[c] - ma[c] * s;
    } else if (idx < 294912 + 32768 + 256 + 96) {
        int c = idx - (294912 + 32768 + 256);
        if (c < 65) {
            float s = gb[c] * rsqrtf(vbv[c] + 1e-5f);
            s2[c] = s;
            b2[c] = bbv[c] * s + btb[c] - mbm[c] * s;
        } else { s2[c] = 0.0f; b2[c] = 0.0f; }
    }
}

// ---------------------------------------------------------------- x -> padded bf16, channel pairs interleaved
// layout: [b][c2(64)][ry(62)][rx(88)][par(2)]  (one u32 per pixel per channel-pair)
__global__ __launch_bounds__(256) void padcvt_k(const float* __restrict__ x,
                                                unsigned short* __restrict__ xp)
{
    int idx = blockIdx.x * 256 + threadIdx.x;
    if (idx >= NB * 64 * XPLANE * 2) return;
    int par = idx & 1;
    int t   = idx >> 1;
    int rx  = t % XPADX;  t /= XPADX;
    int ry  = t % XPADY;  t /= XPADY;
    int c2  = t & 63;
    int b   = t >> 6;
    int sy = ry - 1, sx = rx - 1;
    float v = 0.0f;
    if (sy >= 0 && sy < IH && sx >= 0 && sx < IW)
        v = x[(((size_t)b * CIN + c2 * 2 + par) * IH + sy) * IW + sx];
    xp[idx] = f2bf(v);
}

// ---------------------------------------------------------------- conv1: 3x3 implicit GEMM, BN+ReLU -> h bf16 [b][p][256]
__device__ __forceinline__ void c1_issue(
    int s, const unsigned short* __restrict__ wpk, const unsigned* __restrict__ xp32,
    unsigned short* AsBuf, unsigned short* BsBuf,
    int tid, size_t bBase, const int* spo, const int* ldsB, const int* icpArr)
{
    // A: 256 oc x 32 ic packed weights (64B/thread, contiguous both sides)
    const unsigned short* asrc = wpk + (size_t)s * 8192 + tid * 32;
    unsigned short* adst = AsBuf + tid * 40;
#if USE_ASYNC
    g2l128(asrc,      adst);
    g2l128(asrc + 8,  adst + 8);
    g2l128(asrc + 16, adst + 16);
    g2l128(asrc + 24, adst + 24);
#else
    {
        const uint4* src = (const uint4*)asrc;
        uint4* dst = (uint4*)adst;
        dst[0] = src[0]; dst[1] = src[1]; dst[2] = src[2]; dst[3] = src[3];
    }
#endif
    // B: 64 pixels x 32 ic, transposing gather of channel pairs (4 x b32)
    const int tap = s >> 2, icc = s & 3;
    const int ky = tap / 3 - 1, kx = tap % 3 - 1;
    const int dpix = ky * XPADX + kx;
#pragma unroll
    for (int j = 0; j < 4; ++j) {
        size_t g = bBase + (size_t)(icc * 16 + icpArr[j]) * XPLANE + (spo[j] + dpix);
#if USE_ASYNC
        g2l32(xp32 + g, BsBuf + ldsB[j]);
#else
        *(unsigned*)(BsBuf + ldsB[j]) = xp32[g];
#endif
    }
}

__global__ __launch_bounds__(256) void conv1_wmma(
    const unsigned* __restrict__ xp32, const unsigned short* __restrict__ wpk,
    const float* __restrict__ s1g, const float* __restrict__ b1g,
    unsigned short* __restrict__ h)
{
    __shared__ __align__(16) unsigned short As[2][256 * 40];  // [oc][32 ic], stride 40
    __shared__ __align__(16) unsigned short Bs[2][64 * 40];   // [pixel][32 ic], stride 40
    __shared__ float s1s[256], b1s[256];

    const int tid  = threadIdx.x;
    const int lane = tid & 31, wv = tid >> 5;
    const int wm = wv & 3, wn = wv >> 2;       // wave grid 4(M) x 2(N)
    const int n0 = blockIdx.x * 64;
    const int b  = blockIdx.y;

    s1s[tid] = s1g[tid];
    b1s[tid] = b1g[tid];

    // per-thread B-gather geometry (4 channel-pair elements per stage)
    int spo[4], ldsB[4], icpArr[4];
#pragma unroll
    for (int j = 0; j < 4; ++j) {
        const int idx2 = tid + j * 256;
        const int pl = idx2 & 63;
        icpArr[j] = idx2 >> 6;                 // 0..15 channel pair within chunk
        const int p = n0 + pl;
        const int yy = p / IW, xx2 = p % IW;
        spo[j]  = (yy + 1) * XPADX + (xx2 + 1);
        ldsB[j] = pl * 40 + icpArr[j] * 2;     // short units (4B aligned)
    }
    const size_t bBase = (size_t)b * 64 * XPLANE;

    v8f acc[4][2];
#pragma unroll
    for (int i = 0; i < 4; ++i)
#pragma unroll
        for (int j = 0; j < 2; ++j)
#pragma unroll
            for (int r = 0; r < 8; ++r) acc[i][j][r] = 0.0f;

    c1_issue(0, wpk, xp32, As[0], Bs[0], tid, bBase, spo, ldsB, icpArr);

    for (int s = 0; s < 36; ++s) {
        if (s < 35)
            c1_issue(s + 1, wpk, xp32, As[(s + 1) & 1], Bs[(s + 1) & 1],
                     tid, bBase, spo, ldsB, icpArr);
#if USE_ASYNC
        if (s < 35) __builtin_amdgcn_s_wait_asynccnt(8);   // 8 newest ops may remain
        else        __builtin_amdgcn_s_wait_asynccnt(0);
#endif
        __syncthreads();

        const unsigned short* Ab = As[s & 1];
        const unsigned short* Bb = Bs[s & 1];
        v16bf af[4];
#pragma unroll
        for (int mb = 0; mb < 4; ++mb) {
            const int row  = wm * 64 + mb * 16 + (lane & 15);
            const int hoff = (lane >> 4) * 8;
            v8bf* ap = (v8bf*)&af[mb];
            ap[0] = *(const v8bf*)(Ab + row * 40 + hoff);
            ap[1] = *(const v8bf*)(Ab + row * 40 + hoff + 16);
        }
        v16bf bfr[2];
#pragma unroll
        for (int nb = 0; nb < 2; ++nb) {
            const int nr   = wn * 32 + nb * 16 + (lane & 15);
            const int koff = (lane >> 4) * 8;
            v8bf* bp = (v8bf*)&bfr[nb];
            bp[0] = *(const v8bf*)(Bb + nr * 40 + koff);
            bp[1] = *(const v8bf*)(Bb + nr * 40 + koff + 16);
        }
#pragma unroll
        for (int mb = 0; mb < 4; ++mb)
#pragma unroll
            for (int nb = 0; nb < 2; ++nb)
                acc[mb][nb] = __builtin_amdgcn_wmma_f32_16x16x32_bf16(
                    false, af[mb], false, bfr[nb], (short)0, acc[mb][nb], false, false);
        __syncthreads();
    }

    // epilogue: BN + ReLU, pack 8 bf16 per lane per tile, store h[b][p][c]
#pragma unroll
    for (int mb = 0; mb < 4; ++mb) {
        const int cb = wm * 64 + mb * 16 + (lane >> 4) * 8;
#pragma unroll
        for (int nb = 0; nb < 2; ++nb) {
            const int p = n0 + wn * 32 + nb * 16 + (lane & 15);
            unsigned q[4];
#pragma unroll
            for (int r2 = 0; r2 < 4; ++r2) {
                const int c0 = cb + r2 * 2;
                float v0 = fmaxf(acc[mb][nb][r2 * 2]     * s1s[c0]     + b1s[c0],     0.0f);
                float v1 = fmaxf(acc[mb][nb][r2 * 2 + 1] * s1s[c0 + 1] + b1s[c0 + 1], 0.0f);
                q[r2] = (unsigned)f2bf(v0) | ((unsigned)f2bf(v1) << 16);
            }
            uint4 val; val.x = q[0]; val.y = q[1]; val.z = q[2]; val.w = q[3];
            *(uint4*)(h + (((size_t)b * NP + p) * CHID + cb)) = val;
        }
    }
}

// ---------------------------------------------------------------- conv2: 1x1 GEMM + BN -> logits + pixel-shuffled prob
__device__ __forceinline__ void c2_issue(
    int cc, const unsigned short* __restrict__ w2, const unsigned short* __restrict__ h,
    unsigned short* AsBuf, unsigned short* BsBuf, int tid, int b, int n0)
{
    const int c0 = cc * 32;
    // A: 128 rows x 32 ch = 512 b128-chunks, 2 per thread (uniform)
#pragma unroll
    for (int k = 0; k < 2; ++k) {
        const int c = tid + k * 256;
        const int row = c >> 2, q = c & 3;
        const unsigned short* src = w2 + (size_t)row * 256 + c0 + q * 8;
        unsigned short* dst = AsBuf + row * 40 + q * 8;
#if USE_ASYNC
        g2l128(src, dst);
#else
        *(uint4*)dst = *(const uint4*)src;
#endif
    }
    // B: 64 pixels x 32 ch from h (channel-contiguous), 1 chunk per thread
    {
        const int pl = tid >> 2, qo = tid & 3;
        const unsigned short* src = h + (((size_t)b * NP + n0 + pl) * CHID + c0 + qo * 8);
        unsigned short* dst = BsBuf + pl * 40 + qo * 8;
#if USE_ASYNC
        g2l128(src, dst);
#else
        *(uint4*)dst = *(const uint4*)src;
#endif
    }
}

__global__ __launch_bounds__(256) void conv2_wmma(
    const unsigned short* __restrict__ w2, const float* __restrict__ s2g,
    const float* __restrict__ b2g, const unsigned short* __restrict__ h,
    float* __restrict__ logits, float* __restrict__ prob)
{
    __shared__ __align__(16) unsigned short As[2][128 * 40];
    __shared__ __align__(16) unsigned short Bs[2][64 * 40];
    __shared__ float s2s[96], b2s[96];

    const int tid = threadIdx.x, lane = tid & 31, wv = tid >> 5;
    const int n0 = blockIdx.x * 64, b = blockIdx.y;
    if (tid < 96) { s2s[tid] = s2g[tid]; b2s[tid] = b2g[tid]; }

    v8f acc[3];
#pragma unroll
    for (int q = 0; q < 3; ++q)
#pragma unroll
        for (int r = 0; r < 8; ++r) acc[q][r] = 0.0f;

    c2_issue(0, w2, h, As[0], Bs[0], tid, b, n0);

    for (int cc = 0; cc < 8; ++cc) {
        if (cc < 7) c2_issue(cc + 1, w2, h, As[(cc + 1) & 1], Bs[(cc + 1) & 1], tid, b, n0);
#if USE_ASYNC
        if (cc < 7) __builtin_amdgcn_s_wait_asynccnt(3);   // 3 newest ops may remain
        else        __builtin_amdgcn_s_wait_asynccnt(0);
#endif
        __syncthreads();

        const unsigned short* Ab = As[cc & 1];
        const unsigned short* Bb = Bs[cc & 1];
#pragma unroll
        for (int q = 0; q < 3; ++q) {
            const int id = wv * 3 + q, mt = id >> 2, nt = id & 3;
            v16bf a, bv;
            {
                const int row = mt * 16 + (lane & 15), hoff = (lane >> 4) * 8;
                v8bf* ap = (v8bf*)&a;
                ap[0] = *(const v8bf*)(Ab + row * 40 + hoff);
                ap[1] = *(const v8bf*)(Ab + row * 40 + hoff + 16);
            }
            {
                const int nr = nt * 16 + (lane & 15), koff = (lane >> 4) * 8;
                v8bf* bp = (v8bf*)&bv;
                bp[0] = *(const v8bf*)(Bb + nr * 40 + koff);
                bp[1] = *(const v8bf*)(Bb + nr * 40 + koff + 16);
            }
            acc[q] = __builtin_amdgcn_wmma_f32_16x16x32_bf16(
                false, a, false, bv, (short)0, acc[q], false, false);
        }
        __syncthreads();
    }

#pragma unroll
    for (int q = 0; q < 3; ++q) {
        const int id = wv * 3 + q, mt = id >> 2, nt = id & 3;
        const int cb = mt * 16 + (lane >> 4) * 8;
        const int p  = n0 + nt * 16 + (lane & 15);
        const int yy = p / IW, xx2 = p % IW;
#pragma unroll
        for (int r = 0; r < 8; ++r) {
            const int c = cb + r;
            if (c < 65) {
                float v = acc[q][r] * s2s[c] + b2s[c];
                logits[((size_t)b * 65 + c) * NP + p] = v;
                if (c < 64)
                    prob[(size_t)b * 307200 + (yy * 8 + (c >> 3)) * PW + xx2 * 8 + (c & 7)] = v;
            }
        }
    }
}

// ---------------------------------------------------------------- NMS: separable 9x9 max + equality keep
__global__ __launch_bounds__(256) void hmax_k(const float* __restrict__ prob, float* __restrict__ tmp)
{
    int idx = blockIdx.x * 256 + threadIdx.x;
    if (idx >= NPIX) return;
    int r = idx % (PH * PW);
    int xx = r % PW;
    int base = idx - xx;
    float m = -3.4028235e38f;
    for (int dx = -4; dx <= 4; ++dx) {
        int sx = xx + dx;
        if (sx >= 0 && sx < PW) m = fmaxf(m, prob[base + sx]);
    }
    tmp[idx] = m;
}
__global__ __launch_bounds__(256) void vkeep_k(const float* __restrict__ prob,
                                              const float* __restrict__ tmp,
                                              float* __restrict__ nms)
{
    int idx = blockIdx.x * 256 + threadIdx.x;
    if (idx >= NPIX) return;
    int r = idx % (PH * PW);
    int yy = r / PW, xx = r % PW;
    int b0 = idx - r;
    float m = -3.4028235e38f;
    for (int dy = -4; dy <= 4; ++dy) {
        int sy = yy + dy;
        if (sy >= 0 && sy < PH) m = fmaxf(m, tmp[b0 + sy * PW + xx]);
    }
    float p0 = prob[idx];
    nms[idx] = (p0 == m) ? p0 : 0.0f;
}

// ---------------------------------------------------------------- exact top-k threshold via 2-pass radix select
__global__ __launch_bounds__(256) void hzero_k(unsigned* __restrict__ hist)
{
    int idx = blockIdx.x * 256 + threadIdx.x;
    if (idx < NB * 65536) hist[idx] = 0u;
}
__global__ __launch_bounds__(256) void hfill1_k(const float* __restrict__ nms, unsigned* __restrict__ hist)
{
    int idx = blockIdx.x * 256 + threadIdx.x;
    if (idx >= NPIX) return;
    int b = idx / (PH * PW);
    unsigned u = orderable(nms[idx]);
    atomicAdd(hist + (size_t)b * 65536 + (u >> 16), 1u);
}
__global__ __launch_bounds__(256) void hfill2_k(const float* __restrict__ nms,
                                               const unsigned* __restrict__ selHi,
                                               unsigned* __restrict__ hist)
{
    int idx = blockIdx.x * 256 + threadIdx.x;
    if (idx >= NPIX) return;
    int b = idx / (PH * PW);
    unsigned u = orderable(nms[idx]);
    if ((u >> 16) == selHi[b])
        atomicAdd(hist + (size_t)b * 65536 + (u & 0xFFFFu), 1u);
}
__global__ __launch_bounds__(256) void radix_select(
    const unsigned* __restrict__ hist, const unsigned* __restrict__ kIn, unsigned constK,
    unsigned* __restrict__ selOut, unsigned* __restrict__ kRemOut,
    const unsigned* __restrict__ selHiPrev, float* __restrict__ kthOut)
{
    const int b = blockIdx.x, t = threadIdx.x;
    __shared__ unsigned csum[256];
    const unsigned* hb = hist + (size_t)b * 65536;
    unsigned s = 0;
    for (int j = 0; j < 256; ++j) s += hb[t * 256 + j];
    csum[t] = s;
    __syncthreads();
    if (t == 0) {
        unsigned target = kIn ? kIn[b] : constK;
        if (target == 0) target = 1;
        unsigned cum = 0; int chunk = 0;
        for (int ci = 255; ci >= 0; --ci) {
            unsigned cs = csum[ci];
            if (cum + cs >= target) { chunk = ci; break; }
            cum += cs;
        }
        unsigned rem = target - cum, cum2 = 0; int bin = chunk * 256;
        for (int j = 255; j >= 0; --j) {
            unsigned c = hb[chunk * 256 + j];
            if (cum2 + c >= rem) { bin = chunk * 256 + j; rem = rem - cum2; break; }
            cum2 += c;
        }
        selOut[b]  = (unsigned)bin;
        kRemOut[b] = rem;
        if (kthOut) {
            unsigned u = (selHiPrev[b] << 16) | (unsigned)bin;
            kthOut[b] = fromOrderable(u);
        }
    }
}
__global__ __launch_bounds__(256) void thresh_k(const float* __restrict__ kth,
                                                float* __restrict__ nms,
                                                float* __restrict__ pred)
{
    int idx = blockIdx.x * 256 + threadIdx.x;
    if (idx >= NPIX) return;
    int b = idx / (PH * PW);
    float v = nms[idx];
    float o = (v >= kth[b]) ? v : 0.0f;
    nms[idx]  = o;
    pred[idx] = (o >= 0.015f) ? 1.0f : 0.0f;
}

// ---------------------------------------------------------------- launch
extern "C" void kernel_launch(void* const* d_in, const int* in_sizes, int n_in,
                              void* d_out, int out_size, void* d_ws, size_t ws_size,
                              hipStream_t stream)
{
    const float* x   = (const float*)d_in[0];
    const float* wa  = (const float*)d_in[1];
    const float* ba  = (const float*)d_in[2];
    const float* ga  = (const float*)d_in[3];
    const float* bta = (const float*)d_in[4];
    const float* ma  = (const float*)d_in[5];
    const float* va  = (const float*)d_in[6];
    const float* wb  = (const float*)d_in[7];
    const float* bbv = (const float*)d_in[8];
    const float* gb  = (const float*)d_in[9];
    const float* btb = (const float*)d_in[10];
    const float* mbm = (const float*)d_in[11];
    const float* vbv = (const float*)d_in[12];

    unsigned char* ws = (unsigned char*)d_ws;
    float* out    = (float*)d_out;
    float* logits = out;
    float* prob   = out + LOGITS_N;
    float* nms    = out + LOGITS_N + NPIX;
    float* pred   = out + LOGITS_N + 2 * NPIX;

    unsigned short* wpk1 = (unsigned short*)(ws + WPK1_OFF);
    unsigned short* w2pk = (unsigned short*)(ws + W2PK_OFF);
    float* s1 = (float*)(ws + S1_OFF);
    float* b1 = (float*)(ws + B1_OFF);
    float* s2 = (float*)(ws + S2_OFF);
    float* b2 = (float*)(ws + B2_OFF);
    unsigned* selHi = (unsigned*)(ws + SEL_OFF);
    unsigned* kRem  = (unsigned*)(ws + SEL_OFF + 128);
    unsigned* selLo = (unsigned*)(ws + SEL_OFF + 256);
    unsigned* kRem2 = (unsigned*)(ws + SEL_OFF + 384);
    float*    kth   = (float*)(ws + SEL_OFF + 512);
    unsigned short* hbuf = (unsigned short*)(ws + H_OFF);
    float*    tmp   = (float*)(ws + HMAX_OFF);
    unsigned* hist  = (unsigned*)(ws + HIST_OFF);
    unsigned short* xpad = (unsigned short*)(ws + XPAD_OFF);
    const unsigned* xp32 = (const unsigned*)(ws + XPAD_OFF);

    prep_k<<<1282, 256, 0, stream>>>(wa, ba, ga, bta, ma, va, wb, bbv, gb, btb, mbm, vbv,
                                     wpk1, w2pk, s1, b1, s2, b2);
    padcvt_k<<<(NB * 64 * XPLANE * 2 + 255) / 256, 256, 0, stream>>>(x, xpad);
    conv1_wmma<<<dim3(75, 32), 256, 0, stream>>>(xp32, wpk1, s1, b1, hbuf);
    conv2_wmma<<<dim3(75, 32), 256, 0, stream>>>(w2pk, s2, b2, hbuf, logits, prob);
    hmax_k<<<38400, 256, 0, stream>>>(prob, tmp);
    vkeep_k<<<38400, 256, 0, stream>>>(prob, tmp, nms);
    hzero_k<<<8192, 256, 0, stream>>>(hist);
    hfill1_k<<<38400, 256, 0, stream>>>(nms, hist);
    radix_select<<<32, 256, 0, stream>>>(hist, nullptr, TOPK, selHi, kRem, nullptr, nullptr);
    hzero_k<<<8192, 256, 0, stream>>>(hist);
    hfill2_k<<<38400, 256, 0, stream>>>(nms, selHi, hist);
    radix_select<<<32, 256, 0, stream>>>(hist, kRem, 0u, selLo, kRem2, selHi, kth);
    thresh_k<<<38400, 256, 0, stream>>>(kth, nms, pred);
}
